// WeightSharedBlock_81355270521456
// MI455X (gfx1250) — matrix operand order repaired and checked by reading, not compiled, not run on hardware
//
#include <hip/hip_runtime.h>
#include <hip/hip_bf16.h>

// ---------------------------------------------------------------------------
// MI455X (gfx1250) transformer block.
// Compute-bound (~43 GFLOP vs ~10us of HBM traffic at 23.3TB/s): all GEMMs on
// v_wmma_f32_16x16x32_bf16 (wave32, f32 accum). GEMM tiles are DMA'd into LDS
// by the Tensor Data Mover (TENSORcnt) with double buffering.
// ---------------------------------------------------------------------------

typedef __attribute__((ext_vector_type(16))) __bf16 v16bf;
typedef __attribute__((ext_vector_type(8)))  float  v8f;
typedef __attribute__((ext_vector_type(4)))  unsigned tdm_u4;
typedef __attribute__((ext_vector_type(8)))  int      tdm_i8;
typedef __attribute__((ext_vector_type(4)))  int      tdm_i4;

union ABf { v16bf v; unsigned u[8]; };

__device__ __forceinline__ v8f wmma_bf16(const ABf& a, const ABf& b, v8f c) {
  return __builtin_amdgcn_wmma_f32_16x16x32_bf16(false, a.v, false, b.v,
                                                 (short)0, c, false, false);
}

__device__ __forceinline__ v8f zero8() {
  v8f z; for (int i = 0; i < 8; ++i) z[i] = 0.0f; return z;
}

__device__ __forceinline__ float gelu_exact(float x) {
  return 0.5f * x * (1.0f + erff(x * 0.70710678118654752f));
}

#if __has_builtin(__builtin_amdgcn_tensor_load_to_lds) && \
    __has_builtin(__builtin_amdgcn_s_wait_tensorcnt)
#define USE_TDM 1
#else
#define USE_TDM 0
#endif

#if USE_TDM
// 2D tile DMA: global (rows of tileDim0 bf16, row stride stride0 elems) ->
// LDS rows of tileDim0 bf16 padded to 20 dwords (pad 4 dwords per 16 dwords).
// D# packing per CDNA5 ISA section 8 (group0 128b, group1 256b).
// This toolchain's builtin is the 6-arg form:
//   (uint32x4 g0, int32x8 g1, int32x4 g2, int32x4 g3, int32x8 extra, i32 cpol)
__device__ __forceinline__ void tdm_load_tile(unsigned ldsOff, const void* gptr,
                                              unsigned tensorDim0, unsigned tensorDim1,
                                              unsigned tileDim0, unsigned tileDim1,
                                              unsigned stride0) {
  unsigned long long ga = (unsigned long long)gptr;
  tdm_u4 g0;
  g0[0] = 1u;                                   // count=1, user mode, no gather
  g0[1] = ldsOff;                               // lds_addr (bytes)
  g0[2] = (unsigned)ga;                         // global_addr[31:0]
  g0[3] = (unsigned)(ga >> 32) | (2u << 30);    // global_addr[56:32] | type=2
  tdm_i8 g1;
  g1[0] = (int)((1u << 16) |                    // data_size = 1 (2 bytes)
                (1u << 20) |                    // pad_enable
                (3u << 22) |                    // pad_interval: 16 dwords
                (3u << 25));                    // pad_amount: 4 dwords
  g1[1] = (int)((tensorDim0 & 0xffffu) << 16);  // tensor_dim0[15:0] @ bits 63:48
  g1[2] = (int)((tensorDim0 >> 16) | ((tensorDim1 & 0xffffu) << 16));
  g1[3] = (int)((tensorDim1 >> 16) | (tileDim0 << 16));
  g1[4] = (int)tileDim1;                        // tile_dim1; tile_dim2 = 0
  g1[5] = (int)stride0;                         // tensor_dim0_stride[31:0]
  g1[6] = 0;                                    // stride0 hi, dim1_stride lo
  g1[7] = 0;
  tdm_i4 z4 = {0, 0, 0, 0};
  tdm_i8 z8 = {0, 0, 0, 0, 0, 0, 0, 0};
  __builtin_amdgcn_tensor_load_to_lds(g0, g1, z4, z4, z8, 0);
}
#endif

// ---------------------------------------------------------------------------
// f32 -> bf16 elementwise convert (weights)
// ---------------------------------------------------------------------------
__global__ void cvt_bf16_kernel(const float* __restrict__ src,
                                __hip_bfloat16* __restrict__ dst, int n) {
  int i = blockIdx.x * 256 + threadIdx.x;
  if (i < n) dst[i] = __float2bfloat16(src[i]);
}

// ---------------------------------------------------------------------------
// LayerNorm over D=384, one row per workgroup (128 threads, 3 elems/thread).
// ---------------------------------------------------------------------------
template <bool OUT_BF16>
__global__ __launch_bounds__(128)
void ln_kernel(const float* __restrict__ in, const float* __restrict__ gam,
               const float* __restrict__ bet, void* __restrict__ out) {
  const int row = blockIdx.x;
  const float* x = in + (size_t)row * 384;
  const int tid = threadIdx.x, lane = tid & 31, wid = tid >> 5;
  __shared__ float red[4];

  float v0 = x[tid], v1 = x[tid + 128], v2 = x[tid + 256];
  float s = v0 + v1 + v2;
  for (int o = 16; o > 0; o >>= 1) s += __shfl_xor(s, o);
  if (lane == 0) red[wid] = s;
  __syncthreads();
  const float mean = (red[0] + red[1] + red[2] + red[3]) * (1.0f / 384.0f);

  float d0 = v0 - mean, d1 = v1 - mean, d2 = v2 - mean;
  float q = d0 * d0 + d1 * d1 + d2 * d2;
  for (int o = 16; o > 0; o >>= 1) q += __shfl_xor(q, o);
  __syncthreads();
  if (lane == 0) red[wid] = q;
  __syncthreads();
  const float var = (red[0] + red[1] + red[2] + red[3]) * (1.0f / 384.0f);
  const float inv = rsqrtf(var + 1e-5f);

  float r[3] = {d0, d1, d2};
  for (int j = 0; j < 3; ++j) {
    int d = tid + j * 128;
    float y = r[j] * inv * gam[d] + bet[d];
    if (OUT_BF16) ((__hip_bfloat16*)out)[(size_t)row * 384 + d] = __float2bfloat16(y);
    else          ((float*)out)[(size_t)row * 384 + d] = y;
  }
}

// ---------------------------------------------------------------------------
// Generic bf16 WMMA GEMM: C[M][Nout] = A[M][K] * W[Nout][K]^T (+ epilogue).
// WG tile 256(M) x 64(N), 8 waves as 4(M) x 2(N); each wave owns a 64x32
// register tile (4x2 wmma accumulators, 8 WMMAs per 32-wide K step).
// LDS tiles double-buffered and filled by the Tensor Data Mover.
// MODE 1: QKV  -> scatter q,k per-head row-major; v transposed [d][m] (bf16)
// MODE 2: proj -> +bias +residual(f32) -> f32
// MODE 3: fc1  -> +bias, exact GELU -> bf16
// MODE 4: fc2  -> +bias +residual(f32) -> f32 (final output)
// MODE 6: attn-out (batched over z=(b,h)) -> bf16 packed back to [b,n,D]
// ---------------------------------------------------------------------------
template <int MODE>
__global__ __launch_bounds__(256)
void gemm_wmma(const __hip_bfloat16* __restrict__ A,
               const __hip_bfloat16* __restrict__ W,
               const float* __restrict__ bias,
               const float* __restrict__ resid,
               void* __restrict__ out0, void* __restrict__ out1,
               void* __restrict__ out2,
               int M, int Nout, int K,
               long long aStrideZ, long long wStrideZ) {
  const int tid = threadIdx.x;
  const int lane = tid & 31, wid = tid >> 5;
  const int l16 = lane & 15, lh = (lane >> 4) & 1;
  const int m0 = blockIdx.y * 256;
  const int n0 = blockIdx.x * 64;
  A += (size_t)blockIdx.z * (size_t)aStrideZ;
  W += (size_t)blockIdx.z * (size_t)wStrideZ;

  // Rows of 16 data dwords + 4 pad dwords (matches TDM pad settings).
  __shared__ __align__(16) unsigned sA[2][256][20];
  __shared__ __align__(16) unsigned sW[2][64][20];

  const int mBase = (wid & 3) * 64;   // wave's M sub-tile (64 rows)
  const int nBase = (wid >> 2) * 32;  // wave's N sub-tile (32 cols)

  v8f acc[4][2];
  for (int mi = 0; mi < 4; ++mi)
    for (int ni = 0; ni < 2; ++ni) acc[mi][ni] = zero8();

  const int kTiles = K >> 5;

#if USE_TDM
  const unsigned ldsA[2] = {(unsigned)(unsigned long long)(void*)&sA[0][0][0],
                            (unsigned)(unsigned long long)(void*)&sA[1][0][0]};
  const unsigned ldsW[2] = {(unsigned)(unsigned long long)(void*)&sW[0][0][0],
                            (unsigned)(unsigned long long)(void*)&sW[1][0][0]};
  if (wid == 0) {
    tdm_load_tile(ldsA[0], A + (size_t)m0 * K, K, 256, 32, 256, K);
    tdm_load_tile(ldsW[0], W + (size_t)n0 * K, K, 64, 32, 64, K);
    __builtin_amdgcn_s_wait_tensorcnt(0);
  }
  __syncthreads();
#endif

  for (int kt = 0; kt < kTiles; ++kt) {
    const int cur = kt & 1;
#if USE_TDM
    if (kt + 1 < kTiles && wid == 0) {   // DMA next tiles while we compute
      const size_t kOff = (size_t)(kt + 1) * 32;
      tdm_load_tile(ldsA[cur ^ 1], A + (size_t)m0 * K + kOff, K, 256, 32, 256, K);
      tdm_load_tile(ldsW[cur ^ 1], W + (size_t)n0 * K + kOff, K, 64, 32, 64, K);
    }
#else
    // Synchronous staging fallback (VMEM -> VGPR -> LDS).
    __syncthreads();
    const size_t kOff = (size_t)kt * 32;
    for (int i = tid; i < 4096; i += 256) {
      int r = i >> 4, kp = i & 15;
      sA[0][r][kp] = *(const unsigned*)(A + (size_t)(m0 + r) * K + kOff + kp * 2);
    }
    for (int i = tid; i < 1024; i += 256) {
      int r = i >> 4, kp = i & 15;
      sW[0][r][kp] = *(const unsigned*)(W + (size_t)(n0 + r) * K + kOff + kp * 2);
    }
    __syncthreads();
#endif

    // Fragments per ISA 7.12.2 16-bit layouts (contiguous -> ds_load_b128).
    ABf af[4], bfr[2];
    const int bufA = USE_TDM ? cur : 0;
    for (int mi = 0; mi < 4; ++mi)
      for (int v = 0; v < 8; ++v) {
        int kp = (v < 4 ? 0 : 8) + lh * 4 + (v & 3);
        af[mi].u[v] = sA[bufA][mBase + mi * 16 + l16][kp];
      }
    for (int ni = 0; ni < 2; ++ni)
      for (int v = 0; v < 8; ++v)
        bfr[ni].u[v] = sW[bufA][nBase + ni * 16 + l16][(v < 4 ? 0 : 8) + lh * 4 + (v & 3)];

    for (int mi = 0; mi < 4; ++mi)
      for (int ni = 0; ni < 2; ++ni)
        acc[mi][ni] = wmma_bf16(af[mi], bfr[ni], acc[mi][ni]);

#if USE_TDM
    if (kt + 1 < kTiles) {
      if (wid == 0) __builtin_amdgcn_s_wait_tensorcnt(0);
      __syncthreads();
    }
#endif
  }

  // Epilogue. C/D layout: VGPR v -> row v + 8*lh, lane&15 -> col.
  for (int mi = 0; mi < 4; ++mi)
    for (int ni = 0; ni < 2; ++ni)
      for (int v = 0; v < 8; ++v) {
        const int r = m0 + mBase + mi * 16 + v + lh * 8;
        const int c = n0 + nBase + ni * 16 + l16;
        float val = acc[mi][ni][v];
        if (MODE != 6) val += bias[c];

        if (MODE == 1) {
          int part = c / 384, rem = c % 384;
          int head = rem >> 6, d = rem & 63;
          int b = r >> 10, tok = r & 1023;
          __hip_bfloat16 bv = __float2bfloat16(val);
          size_t rowmaj = (((size_t)(b * 6 + head)) * 1024 + tok) * 64 + d;
          if (part == 0)      ((__hip_bfloat16*)out0)[rowmaj] = bv;
          else if (part == 1) ((__hip_bfloat16*)out1)[rowmaj] = bv;
          else ((__hip_bfloat16*)out2)[(((size_t)(b * 6 + head)) * 64 + d) * 1024 + tok] = bv;
        } else if (MODE == 2 || MODE == 4) {
          ((float*)out0)[(size_t)r * Nout + c] = val + resid[(size_t)r * Nout + c];
        } else if (MODE == 3) {
          ((__hip_bfloat16*)out0)[(size_t)r * (size_t)Nout + c] =
              __float2bfloat16(gelu_exact(val));
        } else if (MODE == 6) {
          int b = blockIdx.z / 6, g = blockIdx.z % 6;
          ((__hip_bfloat16*)out0)[((size_t)b * 1024 + r) * 384 + g * 64 + c] =
              __float2bfloat16(val);
        }
      }
}

// ---------------------------------------------------------------------------
// Scores: one WG per (b, 32x32 score tile); 6 waves = 6 heads, each does
// q(32x64) @ k(32x64)^T via 8 WMMAs, then t_before head-mix + scale in LDS.
// ---------------------------------------------------------------------------
__global__ __launch_bounds__(192)
void scores_kernel(const __hip_bfloat16* __restrict__ qb,
                   const __hip_bfloat16* __restrict__ kb,
                   const float* __restrict__ t_before,
                   __hip_bfloat16* __restrict__ outS) {
  const int mt = blockIdx.x, nt = blockIdx.y, b = blockIdx.z;
  const int tid = threadIdx.x;
  const int lane = tid & 31, h = tid >> 5;
  const int l16 = lane & 15, lh = (lane >> 4) & 1;

  __shared__ float sS[6][32][33];
  __shared__ float stb[36];
  if (tid < 36) stb[tid] = t_before[tid];

  const __hip_bfloat16* qbase = qb + (((size_t)(b * 6 + h)) * 1024 + nt * 32) * 64;
  const __hip_bfloat16* kbase = kb + (((size_t)(b * 6 + h)) * 1024 + mt * 32) * 64;

  v8f acc[2][2] = {{zero8(), zero8()}, {zero8(), zero8()}};
  for (int ks = 0; ks < 2; ++ks) {
    ABf af[2], bfr[2];
    for (int ni = 0; ni < 2; ++ni)
      for (int v = 0; v < 8; ++v) {
        int kk = ks * 32 + (v < 4 ? 0 : 16) + lh * 8 + (v & 3) * 2;
        af[ni].u[v] = *(const unsigned*)(qbase + (size_t)(ni * 16 + l16) * 64 + kk);
      }
    for (int mi = 0; mi < 2; ++mi)
      for (int v = 0; v < 8; ++v) {
        int kk = ks * 32 + (v < 4 ? 0 : 16) + lh * 8 + (v & 3) * 2;
        bfr[mi].u[v] = *(const unsigned*)(kbase + (size_t)(mi * 16 + l16) * 64 + kk);
      }
    for (int ni = 0; ni < 2; ++ni)
      for (int mi = 0; mi < 2; ++mi)
        acc[ni][mi] = wmma_bf16(af[ni], bfr[mi], acc[ni][mi]);
  }
  for (int ni = 0; ni < 2; ++ni)
    for (int mi = 0; mi < 2; ++mi)
      for (int v = 0; v < 8; ++v)
        sS[h][ni * 16 + v + lh * 8][mi * 16 + l16] = acc[ni][mi][v];
  __syncthreads();

  for (int idx = tid; idx < 6 * 32 * 32; idx += 192) {
    int g = idx >> 10, e = idx & 1023, n = e >> 5, m = e & 31;
    float a = 0.0f;
    for (int hh = 0; hh < 6; ++hh) a += stb[g * 6 + hh] * sS[hh][n][m];
    outS[(((size_t)(b * 6 + g)) * 1024 + nt * 32 + n) * 1024 + mt * 32 + m] =
        __float2bfloat16(a * 0.125f);
  }
}

// ---------------------------------------------------------------------------
// Softmax + post-softmax head mix, one WG per (b,n) row.
// ---------------------------------------------------------------------------
__global__ __launch_bounds__(256)
void softmax_mix_kernel(const __hip_bfloat16* __restrict__ scores,
                        const float* __restrict__ t_after,
                        __hip_bfloat16* __restrict__ probs) {
  const int b = blockIdx.x >> 10;
  const int n = blockIdx.x & 1023;
  const int tid = threadIdx.x, lane = tid & 31, wid = tid >> 5;
  __shared__ float sp[6][1024];
  __shared__ float red[8];
  __shared__ float sta[36];
  if (tid < 36) sta[tid] = t_after[tid];

  for (int h = 0; h < 6; ++h) {
    const __hip_bfloat16* row = scores + (((size_t)(b * 6 + h)) * 1024 + n) * 1024;
    float v[4];
    float m = -1e30f;
    for (int j = 0; j < 4; ++j) {
      v[j] = __bfloat162float(row[tid + j * 256]);
      m = fmaxf(m, v[j]);
    }
    for (int o = 16; o > 0; o >>= 1) m = fmaxf(m, __shfl_xor(m, o));
    __syncthreads();
    if (lane == 0) red[wid] = m;
    __syncthreads();
    m = red[0];
    for (int w = 1; w < 8; ++w) m = fmaxf(m, red[w]);
    float s = 0.0f;
    for (int j = 0; j < 4; ++j) { v[j] = __expf(v[j] - m); s += v[j]; }
    for (int o = 16; o > 0; o >>= 1) s += __shfl_xor(s, o);
    __syncthreads();
    if (lane == 0) red[wid] = s;
    __syncthreads();
    s = 0.0f;
    for (int w = 0; w < 8; ++w) s += red[w];
    const float inv = 1.0f / s;
    for (int j = 0; j < 4; ++j) sp[h][tid + j * 256] = v[j] * inv;
  }
  __syncthreads();

  for (int j = 0; j < 4; ++j) {
    int mcol = tid + j * 256;
    float ph[6];
    for (int h = 0; h < 6; ++h) ph[h] = sp[h][mcol];
    for (int g = 0; g < 6; ++g) {
      float a = 0.0f;
      for (int h = 0; h < 6; ++h) a += sta[g * 6 + h] * ph[h];
      probs[(((size_t)(b * 6 + g)) * 1024 + n) * 1024 + mcol] = __float2bfloat16(a);
    }
  }
}

// ---------------------------------------------------------------------------
// Depthwise conv1d over tokens (K=7, same pad) + LayerNorm, fused.
// ---------------------------------------------------------------------------
__global__ __launch_bounds__(128)
void conv_ln_kernel(const float* __restrict__ h2, const float* __restrict__ dww,
                    const float* __restrict__ dwb, const float* __restrict__ lg,
                    const float* __restrict__ lb, __hip_bfloat16* __restrict__ out) {
  const int row = blockIdx.x;
  const int b = row >> 10, n = row & 1023;
  const int tid = threadIdx.x, lane = tid & 31, wid = tid >> 5;
  const float* base = h2 + (size_t)b * 1024 * 384;
  __shared__ float red[4];

  float r[3];
  for (int j = 0; j < 3; ++j) {
    int d = tid + j * 128;
    float acc = dwb[d];
    for (int k = 0; k < 7; ++k) {
      int nn = n + k - 3;
      if (nn >= 0 && nn < 1024) acc += base[(size_t)nn * 384 + d] * dww[d * 7 + k];
    }
    r[j] = acc;
  }
  float s = r[0] + r[1] + r[2];
  for (int o = 16; o > 0; o >>= 1) s += __shfl_xor(s, o);
  if (lane == 0) red[wid] = s;
  __syncthreads();
  const float mean = (red[0] + red[1] + red[2] + red[3]) * (1.0f / 384.0f);
  float q = 0.0f;
  for (int j = 0; j < 3; ++j) { float d = r[j] - mean; q += d * d; }
  for (int o = 16; o > 0; o >>= 1) q += __shfl_xor(q, o);
  __syncthreads();
  if (lane == 0) red[wid] = q;
  __syncthreads();
  const float var = (red[0] + red[1] + red[2] + red[3]) * (1.0f / 384.0f);
  const float inv = rsqrtf(var + 1e-5f);
  for (int j = 0; j < 3; ++j) {
    int d = tid + j * 128;
    out[(size_t)row * 384 + d] =
        __float2bfloat16((r[j] - mean) * inv * lg[d] + lb[d]);
  }
}

// ---------------------------------------------------------------------------
// Host orchestration
// ---------------------------------------------------------------------------
extern "C" void kernel_launch(void* const* d_in, const int* in_sizes, int n_in,
                              void* d_out, int out_size, void* d_ws, size_t ws_size,
                              hipStream_t stream) {
  const float* x        = (const float*)d_in[0];
  const float* qkv_w    = (const float*)d_in[1];
  const float* qkv_b    = (const float*)d_in[2];
  const float* proj_w   = (const float*)d_in[3];
  const float* proj_b   = (const float*)d_in[4];
  const float* t_before = (const float*)d_in[5];
  const float* t_after  = (const float*)d_in[6];
  const float* dw_w     = (const float*)d_in[7];
  const float* dw_b     = (const float*)d_in[8];
  const float* mlp_ln_g = (const float*)d_in[9];
  const float* mlp_ln_b = (const float*)d_in[10];
  const float* norm1_g  = (const float*)d_in[11];
  const float* norm1_b  = (const float*)d_in[12];
  const float* norm2_g  = (const float*)d_in[13];
  const float* norm2_b  = (const float*)d_in[14];
  const float* fc1_w    = (const float*)d_in[15];
  const float* fc1_b    = (const float*)d_in[16];
  const float* fc2_w    = (const float*)d_in[17];
  const float* fc2_b    = (const float*)d_in[18];
  float* out = (float*)d_out;

  char* ws = (char*)d_ws;
  auto alloc = [&](size_t bytes) -> char* {
    char* p = ws;
    ws += (bytes + 255) & ~(size_t)255;
    return p;
  };

  __hip_bfloat16* wq = (__hip_bfloat16*)alloc((size_t)1152 * 384 * 2);
  __hip_bfloat16* wp = (__hip_bfloat16*)alloc((size_t)384 * 384 * 2);
  __hip_bfloat16* w1 = (__hip_bfloat16*)alloc((size_t)1536 * 384 * 2);
  __hip_bfloat16* w2 = (__hip_bfloat16*)alloc((size_t)384 * 1536 * 2);
  __hip_bfloat16* h1 = (__hip_bfloat16*)alloc((size_t)8192 * 384 * 2);
  __hip_bfloat16* qb = (__hip_bfloat16*)alloc((size_t)48 * 1024 * 64 * 2);
  __hip_bfloat16* kb = (__hip_bfloat16*)alloc((size_t)48 * 1024 * 64 * 2);
  __hip_bfloat16* vT = (__hip_bfloat16*)alloc((size_t)48 * 64 * 1024 * 2);
  __hip_bfloat16* ao = (__hip_bfloat16*)alloc((size_t)8192 * 384 * 2);
  float*        xatt = (float*)alloc((size_t)8192 * 384 * 4);
  __hip_bfloat16* hm = (__hip_bfloat16*)alloc((size_t)8192 * 384 * 2);
  char* scoresRegion  = alloc((size_t)48 * 1024 * 1024 * 2);   // bf16 scores
  char* probsRegion   = alloc((size_t)48 * 1024 * 1024 * 2);   // bf16 probs
  __hip_bfloat16* scoresB = (__hip_bfloat16*)scoresRegion;
  __hip_bfloat16* probsB  = (__hip_bfloat16*)probsRegion;
  // Region reuse once attention is consumed:
  __hip_bfloat16* hfc = (__hip_bfloat16*)scoresRegion;  // 8192x1536 bf16
  float*          h2  = (float*)probsRegion;            // 8192x384  f32

  // 1) Weights -> bf16
  cvt_bf16_kernel<<<(1152 * 384 + 255) / 256, 256, 0, stream>>>(qkv_w, wq, 1152 * 384);
  cvt_bf16_kernel<<<(384 * 384 + 255) / 256, 256, 0, stream>>>(proj_w, wp, 384 * 384);
  cvt_bf16_kernel<<<(1536 * 384 + 255) / 256, 256, 0, stream>>>(fc1_w, w1, 1536 * 384);
  cvt_bf16_kernel<<<(384 * 1536 + 255) / 256, 256, 0, stream>>>(fc2_w, w2, 384 * 1536);

  // 2) LN1 -> bf16
  ln_kernel<true><<<8192, 128, 0, stream>>>(x, norm1_g, norm1_b, (void*)h1);

  // 3) QKV GEMM (M=8192, N=1152, K=384)
  gemm_wmma<1><<<dim3(18, 32, 1), 256, 0, stream>>>(
      h1, wq, qkv_b, nullptr, (void*)qb, (void*)kb, (void*)vT,
      8192, 1152, 384, 0, 0);

  // 4) Scores + t_before mix + scale -> bf16
  scores_kernel<<<dim3(32, 32, 8), 192, 0, stream>>>(qb, kb, t_before, scoresB);

  // 5) Softmax + t_after mix -> bf16 probs
  softmax_mix_kernel<<<8192, 256, 0, stream>>>(scoresB, t_after, probsB);

  // 6) attn_out = probs' @ v (batched over 48 (b,h); M=1024, N=64, K=1024)
  gemm_wmma<6><<<dim3(1, 4, 48), 256, 0, stream>>>(
      probsB, vT, nullptr, nullptr, (void*)ao, nullptr, nullptr,
      1024, 64, 1024, (long long)1024 * 1024, (long long)64 * 1024);

  // 7) proj GEMM + bias + residual(x) -> f32 xatt
  gemm_wmma<2><<<dim3(6, 32, 1), 256, 0, stream>>>(
      ao, wp, proj_b, x, (void*)xatt, nullptr, nullptr, 8192, 384, 384, 0, 0);

  // 8) LN2 -> f32 (conv input)
  ln_kernel<false><<<8192, 128, 0, stream>>>(xatt, norm2_g, norm2_b, (void*)h2);

  // 9) depthwise conv (K=7) + mlp LN -> bf16
  conv_ln_kernel<<<8192, 128, 0, stream>>>(h2, dw_w, dw_b, mlp_ln_g, mlp_ln_b, hm);

  // 10) fc1 GEMM + bias + GELU -> bf16 (M=8192, N=1536, K=384)
  gemm_wmma<3><<<dim3(24, 32, 1), 256, 0, stream>>>(
      hm, w1, fc1_b, nullptr, (void*)hfc, nullptr, nullptr, 8192, 1536, 384, 0, 0);

  // 11) fc2 GEMM + bias + residual(xatt) -> f32 final output
  gemm_wmma<4><<<dim3(6, 32, 1), 256, 0, stream>>>(
      hfc, w2, fc2_b, xatt, (void*)out, nullptr, nullptr, 8192, 384, 1536, 0, 0);

  (void)in_sizes; (void)n_in; (void)out_size; (void)ws_size;
}
